// DiscretizedManifoldBlock_23974507446368
// MI455X (gfx1250) — matrix operand order repaired and checked by reading, not compiled, not run on hardware
//
#include <hip/hip_runtime.h>

#define Bb 2
#define Ss 2048
#define Dd 1024
#define Hh 16
#define HDd 64
#define KK 1024
#define LL 4
#define FF 4096
#define NT (Bb * Ss)   // 4096 tokens

typedef __bf16 bf16_t;
typedef bf16_t v16bf __attribute__((ext_vector_type(16)));
typedef bf16_t v8bf  __attribute__((ext_vector_type(8)));
typedef bf16_t v4bf  __attribute__((ext_vector_type(4)));
typedef float  v8f   __attribute__((ext_vector_type(8)));
typedef unsigned int v4u __attribute__((ext_vector_type(4)));
typedef int v8i_ __attribute__((ext_vector_type(8)));
typedef int v4i_ __attribute__((ext_vector_type(4)));

#define GM_BIAS 1
#define GM_GELU 2
#define GM_SILU 4
#define GM_ADD  8

#define ROWP 40   // padded LDS row stride in bf16 elements (80B)

#if defined(__has_builtin)
#if __has_builtin(__builtin_amdgcn_tensor_load_to_lds) && \
    __has_builtin(__builtin_amdgcn_s_wait_tensorcnt)
#define HAVE_TDM 1
#endif
#endif

// Fragment-order helpers: ISA 16-bit A-layout  K = (e<8?0:16) + group*8 + (e&7)
__device__ __forceinline__ int frag_pos(int Kidx) {
    int group = ((Kidx & 15) >= 8) ? 1 : 0;
    int e     = ((Kidx < 16) ? 0 : 8) + (Kidx & 7);
    return group * 16 + e;
}

__device__ __forceinline__ v16bf load_frag(const bf16_t* p) {
    v8bf lo = *(const v8bf*)(p);
    v8bf hi = *(const v8bf*)(p + 8);
    return __builtin_shufflevector(lo, hi, 0, 1, 2, 3, 4, 5, 6, 7,
                                   8, 9, 10, 11, 12, 13, 14, 15);
}

__device__ __forceinline__ v4bf cvt4(float a, float b, float c, float d) {
    v4bf r; r[0] = (bf16_t)a; r[1] = (bf16_t)b; r[2] = (bf16_t)c; r[3] = (bf16_t)d;
    return r;
}

// Build an A fragment from a row of a raw f32 LDS tile (row stride 32 floats).
// Fragment K-runs are contiguous: K = grp*8+0..7 and K = 16+grp*8+0..7.
__device__ __forceinline__ v16bf a_frag_from_f32(const float* rowp, int grp) {
    const float4 f0 = *(const float4*)(rowp + grp * 8);
    const float4 f1 = *(const float4*)(rowp + grp * 8 + 4);
    const float4 f2 = *(const float4*)(rowp + 16 + grp * 8);
    const float4 f3 = *(const float4*)(rowp + 16 + grp * 8 + 4);
    v16bf r;
    r[0]  = (bf16_t)f0.x; r[1]  = (bf16_t)f0.y; r[2]  = (bf16_t)f0.z; r[3]  = (bf16_t)f0.w;
    r[4]  = (bf16_t)f1.x; r[5]  = (bf16_t)f1.y; r[6]  = (bf16_t)f1.z; r[7]  = (bf16_t)f1.w;
    r[8]  = (bf16_t)f2.x; r[9]  = (bf16_t)f2.y; r[10] = (bf16_t)f2.z; r[11] = (bf16_t)f2.w;
    r[12] = (bf16_t)f3.x; r[13] = (bf16_t)f3.y; r[14] = (bf16_t)f3.z; r[15] = (bf16_t)f3.w;
    return r;
}

#ifdef HAVE_TDM
// TDM 2-D tile load: tile_d0 x tile_d1 f32 elements, row stride d0_stride
// (elements), from gaddr into LDS offset lds_off. D# per CDNA5 ISA §8.3/8.4.
// 6-arg builtin form (clang-23 / therock headers).
__device__ __forceinline__ void tdm_load_2d_f32(unsigned lds_off, const void* gaddr,
                                                int tensor_d0, long long d0_stride,
                                                int tile_d0, int tile_d1)
{
    unsigned long long ga = (unsigned long long)gaddr;
    v4u g0;
    g0[0] = 1u;                                              // count=1, user mode
    g0[1] = lds_off;                                         // lds_addr (bytes)
    g0[2] = (unsigned)(ga & 0xFFFFFFFFull);                  // global_addr[31:0]
    g0[3] = (unsigned)((ga >> 32) & 0x01FFFFFFull) | 0x80000000u; // [56:32] | type=2
    const unsigned td1 = 1u << 20;                           // generous dim1 extent
    v8i_ g1;
    g1[0] = 0x00020000;                                      // data_size=4B
    g1[1] = (int)((unsigned)(tensor_d0 & 0xFFFF) << 16);     // tensor_dim0 lo16
    g1[2] = (int)(((unsigned)tensor_d0 >> 16) | ((td1 & 0xFFFFu) << 16));
    g1[3] = (int)(((td1 >> 16) & 0xFFFFu) | ((unsigned)tile_d0 << 16)); // tile_dim0
    g1[4] = (int)((unsigned)tile_d1);                        // tile_dim1, tile_dim2=0
    g1[5] = (int)(d0_stride & 0xFFFFFFFFll);                 // dim0_stride lo32
    g1[6] = (int)((d0_stride >> 32) & 0xFFFFll);             // dim0_stride hi16
    g1[7] = 0;
    v4i_ z4; z4[0] = 0; z4[1] = 0; z4[2] = 0; z4[3] = 0;
    v8i_ z8;
    for (int i = 0; i < 8; ++i) z8[i] = 0;
    __builtin_amdgcn_tensor_load_to_lds(g0, g1, z4, z4, z8, 0);
}
#endif

// ---------------------------------------------------------------------------
// bf16-WMMA GEMM. A tile DMA'd raw (TDM) into f32 LDS; B tile staged to LDS in
// WMMA fragment order as bf16. Block = 128 threads (4 waves), tile 64x64, KC 32.
// transB: Bw is [N,Kd] row-major.
// ---------------------------------------------------------------------------
__global__ __launch_bounds__(128)
void gemm_bf16_wmma(const float* __restrict__ A, const float* __restrict__ Bw,
                    float* __restrict__ C, const float* __restrict__ bias,
                    const float* __restrict__ addp,
                    int M, int N, int Kd, int transB, int mode)
{
    __shared__ alignas(16) float  Asf[64 * 32];    // raw f32 A tile (TDM dest)
    __shared__ alignas(16) bf16_t Bs[64 * ROWP];   // col n: 32 frag-ordered bf16
    const int tid  = threadIdx.x;
    const int wave = tid >> 5;
    const int lane = tid & 31;
    const int grp  = lane >> 4;
    const int l15  = lane & 15;
    const int m0 = blockIdx.x * 64;
    const int n0 = blockIdx.y * 64;

    v8f acc[4];
    for (int i = 0; i < 4; ++i)
        for (int e = 0; e < 8; ++e) acc[i][e] = 0.0f;

    for (int kc = 0; kc < Kd; kc += 32) {
#ifdef HAVE_TDM
        if (wave == 0)
            tdm_load_2d_f32((unsigned)(unsigned long long)(&Asf[0]),
                            A + (size_t)m0 * Kd + kc, Kd, (long long)Kd, 32, 64);
#else
        for (int qd = tid; qd < 512; qd += 128) {
            int r = qd >> 3, c = (qd & 7) * 4;
            *(float4*)(&Asf[r * 32 + c]) =
                *(const float4*)(A + (size_t)(m0 + r) * Kd + kc + c);
        }
#endif
        // ---- stage B 32x64 -> fragment order per column (bf16) ----
        for (int qd = tid; qd < 512; qd += 128) {
            int n  = qd & 63;
            int eb = qd >> 6;
            int g  = eb >> 2;
            int e0 = (eb & 3) * 4;
            int K0 = (e0 < 8) ? g * 8 + e0 : 16 + g * 8 + (e0 - 8);
            float a0, a1, a2, a3;
            if (transB) {
                const float4 f = *(const float4*)(Bw + (size_t)(n0 + n) * Kd + kc + K0);
                a0 = f.x; a1 = f.y; a2 = f.z; a3 = f.w;
            } else {
                const float* bp = Bw + (size_t)(kc + K0) * N + n0 + n;
                a0 = bp[0]; a1 = bp[(size_t)N]; a2 = bp[2 * (size_t)N]; a3 = bp[3 * (size_t)N];
            }
            *(v4bf*)(&Bs[n * ROWP + g * 16 + e0]) = cvt4(a0, a1, a2, a3);
        }
#ifdef HAVE_TDM
        __builtin_amdgcn_s_wait_tensorcnt(0);  // wave0's DMA; no-op for others
#endif
        __syncthreads();

        // ---- fragments; load all B frags first so waits collapse ----
        const v16bf af = a_frag_from_f32(&Asf[(wave * 16 + l15) * 32], grp);
        v16bf bfr[4];
#pragma unroll
        for (int ns = 0; ns < 4; ++ns)
            bfr[ns] = load_frag(&Bs[(ns * 16 + l15) * ROWP + grp * 16]);
#pragma unroll
        for (int ns = 0; ns < 4; ++ns)
            acc[ns] = __builtin_amdgcn_wmma_f32_16x16x32_bf16(
                false, af, false, bfr[ns], (short)0, acc[ns], false, false);
        __syncthreads();
    }

    // epilogue + store; C layout: M=(lane>>4)*8+e, N=lane&15
    for (int ns = 0; ns < 4; ++ns) {
        for (int e = 0; e < 8; ++e) {
            int m = m0 + wave * 16 + grp * 8 + e;
            int n = n0 + ns * 16 + l15;
            float v = acc[ns][e];
            if (mode & GM_BIAS) v += bias[n];
            if (mode & GM_GELU) v = 0.5f * v * (1.0f + erff(v * 0.70710678f));
            if (mode & GM_SILU) v = v / (1.0f + expf(-v));
            size_t o = (size_t)m * N + n;
            if (mode & GM_ADD) v += addp[o];
            C[o] = v;
        }
    }
}

// ---------------------------------------------------------------------------
__global__ __launch_bounds__(256)
void ln_kernel(const float* __restrict__ x, const float* __restrict__ g,
               const float* __restrict__ b, float* __restrict__ out)
{
    __shared__ float rs[256], rs2[256];
    const int row = blockIdx.x, tid = threadIdx.x;
    const float* xr = x + (size_t)row * Dd;
    float s = 0.f, s2 = 0.f;
    for (int i = tid; i < Dd; i += 256) { float v = xr[i]; s += v; s2 += v * v; }
    rs[tid] = s; rs2[tid] = s2; __syncthreads();
    for (int st = 128; st > 0; st >>= 1) {
        if (tid < st) { rs[tid] += rs[tid + st]; rs2[tid] += rs2[tid + st]; }
        __syncthreads();
    }
    float mean = rs[0] * (1.0f / Dd);
    float var  = rs2[0] * (1.0f / Dd) - mean * mean;
    float rstd = rsqrtf(var + 1e-5f);
    float* orow = out + (size_t)row * Dd;
    for (int i = tid; i < Dd; i += 256)
        orow[i] = (xr[i] - mean) * rstd * g[i] + b[i];
}

// ---------------------------------------------------------------------------
__global__ void rope_kernel(float* __restrict__ buf)
{
    int i = blockIdx.x * blockDim.x + threadIdx.x;   // over B*S*H*32
    if (i >= Bb * Ss * Hh * 32) return;
    int hd1  = i & 31;
    int rest = i >> 5;
    size_t off = (size_t)rest * 64 + hd1;
    int s = (int)((off / (size_t)Dd) % Ss);
    float inv = powf(10000.0f, -(float)hd1 / 32.0f);
    float f = (float)s * inv;
    float c = cosf(f), sn = sinf(f);
    float x1 = buf[off], x2 = buf[off + 32];
    buf[off]      = x1 * c - x2 * sn;
    buf[off + 32] = x2 * c + x1 * sn;
}

// ---------------------------------------------------------------------------
// Fused causal retention with per-head exponential decay (flash-style).
// ---------------------------------------------------------------------------
__global__ __launch_bounds__(32)
void retention_kernel(const float* __restrict__ q, const float* __restrict__ k,
                      const float* __restrict__ v, float* __restrict__ o)
{
    __shared__ alignas(16) bf16_t sst[16 * ROWP];
    const int lane = threadIdx.x;
    const int grp  = lane >> 4;
    const int l15  = lane & 15;
    const int QT = Ss / 16;
    int blk = blockIdx.x;
    int qt = blk % QT;
    int h  = (blk / QT) % Hh;
    int b  = blk / (QT * Hh);

    float gamma = 1.0f - exp2f(-5.0f - (float)h);
    float lg = log2f(gamma);

    v8f acc[4];
    for (int i = 0; i < 4; ++i)
        for (int e = 0; e < 8; ++e) acc[i][e] = 0.f;

    // Q fragments (pre-scaled by 1/sqrt(HD))
    v16bf qf[2];
    {
        const float* qrow = q + ((size_t)(b * Ss + qt * 16 + l15)) * Dd + h * HDd;
        const int kb = grp * 8;
#pragma unroll
        for (int c2 = 0; c2 < 2; ++c2) {
            const float4 f0 = *(const float4*)(qrow + c2 * 32 + kb);
            const float4 f1 = *(const float4*)(qrow + c2 * 32 + kb + 4);
            const float4 f2 = *(const float4*)(qrow + c2 * 32 + 16 + kb);
            const float4 f3 = *(const float4*)(qrow + c2 * 32 + 16 + kb + 4);
            const float sc8 = 0.125f;
            qf[c2][0]  = (bf16_t)(f0.x * sc8); qf[c2][1]  = (bf16_t)(f0.y * sc8);
            qf[c2][2]  = (bf16_t)(f0.z * sc8); qf[c2][3]  = (bf16_t)(f0.w * sc8);
            qf[c2][4]  = (bf16_t)(f1.x * sc8); qf[c2][5]  = (bf16_t)(f1.y * sc8);
            qf[c2][6]  = (bf16_t)(f1.z * sc8); qf[c2][7]  = (bf16_t)(f1.w * sc8);
            qf[c2][8]  = (bf16_t)(f2.x * sc8); qf[c2][9]  = (bf16_t)(f2.y * sc8);
            qf[c2][10] = (bf16_t)(f2.z * sc8); qf[c2][11] = (bf16_t)(f2.w * sc8);
            qf[c2][12] = (bf16_t)(f3.x * sc8); qf[c2][13] = (bf16_t)(f3.y * sc8);
            qf[c2][14] = (bf16_t)(f3.z * sc8); qf[c2][15] = (bf16_t)(f3.w * sc8);
        }
    }

    const int nch = (qt * 16 + 16 + 31) / 32;   // causal chunk count
    for (int ch = 0; ch < nch; ++ch) {
        const int j = ch * 32;
        v8f sc[2];
        for (int i = 0; i < 2; ++i)
            for (int e = 0; e < 8; ++e) sc[i][e] = 0.f;
#pragma unroll
        for (int sub = 0; sub < 2; ++sub) {
            int key  = j + sub * 16 + l15;
            int keyc = key < Ss ? key : Ss - 1;
            const float* krow = k + ((size_t)(b * Ss + keyc)) * Dd + h * HDd;
#pragma unroll
            for (int c2 = 0; c2 < 2; ++c2) {
                v16bf kf;
                const float* kp = krow + c2 * 32 + grp * 8;
#pragma unroll
                for (int qd4 = 0; qd4 < 4; ++qd4) {
                    const float4 f = *(const float4*)(kp + (qd4 & 1) * 4 + (qd4 >> 1) * 16);
                    kf[qd4 * 4 + 0] = (bf16_t)f.x; kf[qd4 * 4 + 1] = (bf16_t)f.y;
                    kf[qd4 * 4 + 2] = (bf16_t)f.z; kf[qd4 * 4 + 3] = (bf16_t)f.w;
                }
                sc[sub] = __builtin_amdgcn_wmma_f32_16x16x32_bf16(
                    false, qf[c2], false, kf, (short)0, sc[sub], false, false);
            }
        }
        // decay mask; stage scores into LDS in contiguous K order
#pragma unroll
        for (int sub = 0; sub < 2; ++sub) {
            int n    = j + sub * 16 + l15;
            int fp   = sub * 16 + l15;
#pragma unroll
            for (int e = 0; e < 8; ++e) {
                int m = qt * 16 + grp * 8 + e;
                int d = m - n;
                float w = (d >= 0) ? exp2f(lg * (float)d) : 0.0f;
                sst[(grp * 8 + e) * ROWP + fp] = (bf16_t)(sc[sub][e] * w);
            }
        }
        __syncthreads();
        const v16bf af = load_frag(&sst[l15 * ROWP + grp * 16]);
#pragma unroll
        for (int ns = 0; ns < 4; ++ns) {
            v16bf vf;
            const int hd = ns * 16 + l15;
            const int kb = grp * 16;
#pragma unroll
            for (int e = 0; e < 16; ++e) {
                int kr  = j + kb + e;
                int krc = kr < Ss ? kr : Ss - 1;
                vf[e] = (bf16_t)v[((size_t)(b * Ss + krc)) * Dd + h * HDd + hd];
            }
            acc[ns] = __builtin_amdgcn_wmma_f32_16x16x32_bf16(
                false, af, false, vf, (short)0, acc[ns], false, false);
        }
        __syncthreads();
    }
    for (int ns = 0; ns < 4; ++ns)
        for (int e = 0; e < 8; ++e) {
            int m  = qt * 16 + grp * 8 + e;
            int hd = ns * 16 + l15;
            o[((size_t)(b * Ss + m)) * Dd + h * HDd + hd] = acc[ns][e];
        }
}

__global__ void mul_kernel(const float* __restrict__ a, const float* __restrict__ b,
                           float* __restrict__ c, int n)
{
    int i = blockIdx.x * blockDim.x + threadIdx.x;
    if (i < n) c[i] = a[i] * b[i];
}

// ---------------------------------------------------------------------------
__global__ __launch_bounds__(256)
void codenorm_kernel(const float* __restrict__ cb, float* __restrict__ cn)
{
    __shared__ float rs[256];
    int code = blockIdx.x;
    int tid  = threadIdx.x;
    const float* row = cb + (size_t)code * Dd;
    float s = 0.f;
    for (int i = tid; i < Dd; i += 256) { float v = row[i]; s += v * v; }
    rs[tid] = s; __syncthreads();
    for (int st = 128; st > 0; st >>= 1) {
        if (tid < st) rs[tid] += rs[tid + st];
        __syncthreads();
    }
    if (tid == 0) cn[code] = rs[0];
}

__global__ __launch_bounds__(256)
void argmin_kernel(const float* __restrict__ dot, const float* __restrict__ cn,
                   int* __restrict__ idx)
{
    __shared__ float bv[256];
    __shared__ int   bi[256];
    int t = blockIdx.x, tid = threadIdx.x;
    float best = 3.4e38f; int bc = 0;
    for (int c = tid; c < KK; c += 256) {
        float d2 = cn[c] - 2.0f * dot[(size_t)t * KK + c];
        if (d2 < best) { best = d2; bc = c; }
    }
    bv[tid] = best; bi[tid] = bc; __syncthreads();
    for (int st = 128; st > 0; st >>= 1) {
        if (tid < st) {
            if (bv[tid + st] < bv[tid] ||
                (bv[tid + st] == bv[tid] && bi[tid + st] < bi[tid])) {
                bv[tid] = bv[tid + st]; bi[tid] = bi[tid + st];
            }
        }
        __syncthreads();
    }
    if (tid == 0) idx[t] = bi[0];
}

__global__ __launch_bounds__(256)
void rvq_update_kernel(float* __restrict__ residual, const float* __restrict__ cb,
                       const int* __restrict__ idx, float* __restrict__ total,
                       float* __restrict__ sq)
{
    __shared__ float rs[256];
    int t = blockIdx.x, tid = threadIdx.x;
    const float* crow = cb + (size_t)idx[t] * Dd;
    float s = 0.f;
    for (int i = tid; i < Dd; i += 256) {
        float qv = crow[i];
        size_t o = (size_t)t * Dd + i;
        float r = residual[o];
        float d = qv - r;
        s += d * d;
        total[o] += qv;
        residual[o] = r - qv;
    }
    rs[tid] = s; __syncthreads();
    for (int st = 128; st > 0; st >>= 1) {
        if (tid < st) rs[tid] += rs[tid + st];
        __syncthreads();
    }
    if (tid == 0) atomicAdd(sq, rs[0]);
}

__global__ void finalize_kernel(const float* __restrict__ x2, const float* __restrict__ total,
                                const float* __restrict__ sq, float* __restrict__ out)
{
    int i = blockIdx.x * blockDim.x + threadIdx.x;
    const int n = NT * Dd;
    if (i < n) out[i] = x2[i] + total[i];
    if (i == 0) {
        float qv = sq[0] / (float)(NT * Dd) / (float)LL;
        out[n] = qv;
        out[n + 1] = qv;
    }
}

// ---------------------------------------------------------------------------
extern "C" void kernel_launch(void* const* d_in, const int* in_sizes, int n_in,
                              void* d_out, int out_size, void* d_ws, size_t ws_size,
                              hipStream_t stream)
{
    (void)in_sizes; (void)n_in; (void)out_size; (void)ws_size;
    const float* x    = (const float*)d_in[0];
    const float* ln1g = (const float*)d_in[1];
    const float* ln1b = (const float*)d_in[2];
    const float* wq   = (const float*)d_in[3];
    const float* wk   = (const float*)d_in[4];
    const float* wv   = (const float*)d_in[5];
    const float* wg   = (const float*)d_in[6];
    const float* wo   = (const float*)d_in[7];
    const float* ln2g = (const float*)d_in[8];
    const float* ln2b = (const float*)d_in[9];
    const float* w1   = (const float*)d_in[10];
    const float* b1   = (const float*)d_in[11];
    const float* w2   = (const float*)d_in[12];
    const float* b2   = (const float*)d_in[13];
    const float* ln3g = (const float*)d_in[14];
    const float* ln3b = (const float*)d_in[15];
    const float* cb   = (const float*)d_in[16];
    float* out = (float*)d_out;

    float* ws = (float*)d_ws;
    const size_t TD = (size_t)NT * Dd;
    float* h     = ws; ws += TD;
    float* qb    = ws; ws += TD;
    float* kb    = ws; ws += TD;
    float* vb    = ws; ws += TD;
    float* gb    = ws; ws += TD;
    float* ob    = ws; ws += TD;
    float* og    = ws; ws += TD;
    float* x1    = ws; ws += TD;
    float* h2    = ws; ws += TD;
    float* mid   = ws; ws += (size_t)NT * FF;
    float* x2    = ws; ws += TD;
    float* resid = ws; ws += TD;
    float* tot   = ws; ws += TD;
    float* dotb  = ws; ws += (size_t)NT * KK;
    float* cn    = ws; ws += (size_t)LL * KK;
    float* sq    = ws; ws += 64;
    int*   idx   = (int*)ws;

    const dim3 blk128(128);
    const dim3 g1(NT / 64, Dd / 64);
    const int  nel = NT * Dd;

    // --- retention sub-block ---
    ln_kernel<<<NT, 256, 0, stream>>>(x, ln1g, ln1b, h);
    gemm_bf16_wmma<<<g1, blk128, 0, stream>>>(h, wq, qb, nullptr, nullptr, NT, Dd, Dd, 0, 0);
    gemm_bf16_wmma<<<g1, blk128, 0, stream>>>(h, wk, kb, nullptr, nullptr, NT, Dd, Dd, 0, 0);
    gemm_bf16_wmma<<<g1, blk128, 0, stream>>>(h, wv, vb, nullptr, nullptr, NT, Dd, Dd, 0, 0);
    gemm_bf16_wmma<<<g1, blk128, 0, stream>>>(h, wg, gb, nullptr, nullptr, NT, Dd, Dd, 0, GM_SILU);
    const int nrope = Bb * Ss * Hh * 32;
    rope_kernel<<<(nrope + 255) / 256, 256, 0, stream>>>(qb);
    rope_kernel<<<(nrope + 255) / 256, 256, 0, stream>>>(kb);
    retention_kernel<<<Bb * Hh * (Ss / 16), 32, 0, stream>>>(qb, kb, vb, ob);
    mul_kernel<<<(nel + 255) / 256, 256, 0, stream>>>(ob, gb, og, nel);
    gemm_bf16_wmma<<<g1, blk128, 0, stream>>>(og, wo, x1, nullptr, x, NT, Dd, Dd, 0, GM_ADD);

    // --- MLP sub-block ---
    ln_kernel<<<NT, 256, 0, stream>>>(x1, ln2g, ln2b, h2);
    const dim3 g2(NT / 64, FF / 64);
    gemm_bf16_wmma<<<g2, blk128, 0, stream>>>(h2, w1, mid, b1, nullptr, NT, FF, Dd, 0, GM_BIAS | GM_GELU);
    gemm_bf16_wmma<<<g1, blk128, 0, stream>>>(mid, w2, x2, b2, x1, NT, Dd, FF, 0, GM_BIAS | GM_ADD);

    // --- residual VQ ---
    ln_kernel<<<NT, 256, 0, stream>>>(x2, ln3g, ln3b, resid);
    (void)hipMemsetAsync(tot, 0, TD * sizeof(float), stream);
    (void)hipMemsetAsync(sq, 0, sizeof(float), stream);
    codenorm_kernel<<<LL * KK, 256, 0, stream>>>(cb, cn);
    for (int lvl = 0; lvl < LL; ++lvl) {
        const float* cbl = cb + (size_t)lvl * KK * Dd;
        gemm_bf16_wmma<<<dim3(NT / 64, KK / 64), blk128, 0, stream>>>(
            resid, cbl, dotb, nullptr, nullptr, NT, KK, Dd, 1, 0);
        argmin_kernel<<<NT, 256, 0, stream>>>(dotb, cn + (size_t)lvl * KK, idx);
        rvq_update_kernel<<<NT, 256, 0, stream>>>(resid, cbl, idx, tot, sq);
    }
    finalize_kernel<<<(nel + 2 + 255) / 256, 256, 0, stream>>>(x2, tot, sq, out);
}
